// HierarchicalSemanticDecoder_38010460569877
// MI455X (gfx1250) — compile-verified, matrix operand order checked by reading, not run on hardware
//
#include <hip/hip_runtime.h>
#include <hip/hip_bf16.h>
#include <math.h>

// ---------------------------------------------------------------------------
// CDNA5 (gfx1250): GEMM-shaped work on v_wmma_f32_16x16x32_f16 (wave32, f32
// accumulate) with vectorized b128 fragment loads; serial selective scan on
// VALU with LDS-staged B/C vectors. Channel-last layout throughout.
// ---------------------------------------------------------------------------

typedef __attribute__((ext_vector_type(16))) _Float16 v16h;
typedef __attribute__((ext_vector_type(8)))  float    v8f;

enum { ACT_NONE = 0, ACT_RELU = 1, ACT_GELU = 2, ACT_SILU = 3, ACT_SOFTPLUS = 4 };

__device__ __forceinline__ float hsd_act(float x, int act) {
  if (act == ACT_RELU) return x > 0.f ? x : 0.f;
  if (act == ACT_GELU) {                       // tanh approximation (jax default)
    float x3 = x * x * x;
    return 0.5f * x * (1.f + tanhf(0.7978845608028654f * (x + 0.044715f * x3)));
  }
  if (act == ACT_SILU) return x * (1.f / (1.f + __expf(-x)));
  if (act == ACT_SOFTPLUS) return (x > 20.f) ? x : log1pf(__expf(x));
  return x;
}

// Convert 16 f32 (4x float4) to a v16h fragment.
__device__ __forceinline__ v16h hsd_cvt16(float4 a0, float4 a1, float4 b0,
                                          float4 b1) {
  v16h f;
  f[0] = (_Float16)a0.x;  f[1] = (_Float16)a0.y;
  f[2] = (_Float16)a0.z;  f[3] = (_Float16)a0.w;
  f[4] = (_Float16)a1.x;  f[5] = (_Float16)a1.y;
  f[6] = (_Float16)a1.z;  f[7] = (_Float16)a1.w;
  f[8] = (_Float16)b0.x;  f[9] = (_Float16)b0.y;
  f[10] = (_Float16)b0.z; f[11] = (_Float16)b0.w;
  f[12] = (_Float16)b1.x; f[13] = (_Float16)b1.y;
  f[14] = (_Float16)b1.z; f[15] = (_Float16)b1.w;
  return f;
}

// Fast fragment load: p = row base (+k0), halves 0..7 = p[kg..kg+7],
// halves 8..15 = p[16+kg..16+kg+7]. 32B-aligned b128 loads, branch-free.
__device__ __forceinline__ v16h hsd_load_fast(const float* __restrict__ p,
                                              int kg) {
  const float4* q0 = (const float4*)(p + kg);
  const float4* q1 = (const float4*)(p + 16 + kg);
  return hsd_cvt16(q0[0], q0[1], q1[0], q1[1]);
}

// Fast load with predicated zero (select, not branch). p must be readable.
__device__ __forceinline__ v16h hsd_load_fast_pred(const float* __restrict__ p,
                                                   int kg, bool valid) {
  v16h f = hsd_load_fast(p, kg);
#pragma unroll
  for (int h = 0; h < 16; ++h) f[h] = valid ? f[h] : (_Float16)0.f;
  return f;
}

// Masked (edge-tile) load: clamp indices in-range, zero via select. No branches.
__device__ __forceinline__ v16h hsd_load_masked(const float* __restrict__ base,
                                                int ld, int r0, int rmax, int k0,
                                                int kmax, int lane) {
  int r = r0 + (lane & 15);
  bool rok = r < rmax;
  const float* p = base + (size_t)(rok ? r : 0) * (size_t)ld;
  int kg = (lane < 16) ? 0 : 8;
  v16h f;
#pragma unroll
  for (int h = 0; h < 16; ++h) {
    int k = k0 + ((h < 8) ? (kg + h) : (16 + kg + (h - 8)));
    float v = p[(k < kmax) ? k : 0];
    f[h] = (rok && k < kmax) ? (_Float16)v : (_Float16)0.f;
  }
  return f;
}

// ---------------------------------------------------------------------------
// General GEMM (one wave = 16x16 tile), uniform per-block fast/masked switch.
// Y[T,N] = act((X[T,K] * W[N,K]^T + bias) * scale*mul + shift) (+ residual)
// ---------------------------------------------------------------------------
__global__ void hsd_gemm_nt1_kernel(const float* __restrict__ X, int ldx,
                                    const float* __restrict__ Wt, int ldw,
                                    float* __restrict__ Y, int ldy,
                                    int T, int N, int K,
                                    const float* __restrict__ bias,
                                    const float* __restrict__ scale,
                                    float scale_mul,
                                    const float* __restrict__ shift,
                                    const float* __restrict__ res, int ldr,
                                    int act) {
  int lane = threadIdx.x;
  int n0 = blockIdx.x * 16;
  int t0 = blockIdx.y * 16;
  int kg = (lane < 16) ? 0 : 8;
  v8f acc = {};
  bool edge = ((K & 31) != 0) || (n0 + 16 > N) || (t0 + 16 > T);  // uniform
  if (!edge) {
    const float* ap = X + (size_t)(t0 + (lane & 15)) * (size_t)ldx;
    const float* bp = Wt + (size_t)(n0 + (lane & 15)) * (size_t)ldw;
    for (int k0 = 0; k0 < K; k0 += 32) {
      v16h a = hsd_load_fast(ap + k0, kg);
      v16h b = hsd_load_fast(bp + k0, kg);
      acc = __builtin_amdgcn_wmma_f32_16x16x32_f16(false, a, false, b, (short)0,
                                                   acc, false, false);
    }
  } else {
    for (int k0 = 0; k0 < K; k0 += 32) {
      v16h a = hsd_load_masked(X, ldx, t0, T, k0, K, lane);
      v16h b = hsd_load_masked(Wt, ldw, n0, N, k0, K, lane);
      acc = __builtin_amdgcn_wmma_f32_16x16x32_f16(false, a, false, b, (short)0,
                                                   acc, false, false);
    }
  }
  int n = n0 + (lane & 15);
  if (n >= N) return;
  float bi = bias ? bias[n] : 0.f;
  float sc = scale ? scale[n] * scale_mul : 1.f;
  float sh = shift ? shift[n] : 0.f;
  int mb = (lane < 16) ? 0 : 8;
#pragma unroll
  for (int v = 0; v < 8; ++v) {
    long m = (long)t0 + mb + v;
    if (m >= T) continue;
    float xv = (acc[v] + bi) * sc + sh;
    xv = hsd_act(xv, act);
    if (res) xv += res[(size_t)m * ldr + n];
    Y[(size_t)m * ldy + n] = xv;
  }
}

// ---------------------------------------------------------------------------
// Big-tile GEMM: one wave = 32x64 output (2 A frags x 4 B frags -> 8 WMMA per
// K-step). Requires T%32==0, N%64==0, K%32==0 (checked by host).
// ---------------------------------------------------------------------------
__global__ void hsd_gemm_big_kernel(const float* __restrict__ X, int ldx,
                                    const float* __restrict__ Wt, int ldw,
                                    float* __restrict__ Y, int ldy,
                                    int T, int N, int K,
                                    const float* __restrict__ bias,
                                    const float* __restrict__ scale,
                                    float scale_mul,
                                    const float* __restrict__ shift,
                                    const float* __restrict__ res, int ldr,
                                    int act) {
  int lane = threadIdx.x;
  int m = lane & 15;
  int kg = (lane < 16) ? 0 : 8;
  int n0 = blockIdx.x * 64;
  int t0 = blockIdx.y * 32;
  const float* ap0 = X + (size_t)(t0 + m) * (size_t)ldx;
  const float* ap1 = X + (size_t)(t0 + 16 + m) * (size_t)ldx;
  const float* bp0 = Wt + (size_t)(n0 + m) * (size_t)ldw;
  const float* bp1 = Wt + (size_t)(n0 + 16 + m) * (size_t)ldw;
  const float* bp2 = Wt + (size_t)(n0 + 32 + m) * (size_t)ldw;
  const float* bp3 = Wt + (size_t)(n0 + 48 + m) * (size_t)ldw;
  v8f acc[2][4] = {};
  for (int k0 = 0; k0 < K; k0 += 32) {
    v16h a0 = hsd_load_fast(ap0 + k0, kg);
    v16h a1 = hsd_load_fast(ap1 + k0, kg);
    v16h b;
    b = hsd_load_fast(bp0 + k0, kg);
    acc[0][0] = __builtin_amdgcn_wmma_f32_16x16x32_f16(false, a0, false, b, (short)0, acc[0][0], false, false);
    acc[1][0] = __builtin_amdgcn_wmma_f32_16x16x32_f16(false, a1, false, b, (short)0, acc[1][0], false, false);
    b = hsd_load_fast(bp1 + k0, kg);
    acc[0][1] = __builtin_amdgcn_wmma_f32_16x16x32_f16(false, a0, false, b, (short)0, acc[0][1], false, false);
    acc[1][1] = __builtin_amdgcn_wmma_f32_16x16x32_f16(false, a1, false, b, (short)0, acc[1][1], false, false);
    b = hsd_load_fast(bp2 + k0, kg);
    acc[0][2] = __builtin_amdgcn_wmma_f32_16x16x32_f16(false, a0, false, b, (short)0, acc[0][2], false, false);
    acc[1][2] = __builtin_amdgcn_wmma_f32_16x16x32_f16(false, a1, false, b, (short)0, acc[1][2], false, false);
    b = hsd_load_fast(bp3 + k0, kg);
    acc[0][3] = __builtin_amdgcn_wmma_f32_16x16x32_f16(false, a0, false, b, (short)0, acc[0][3], false, false);
    acc[1][3] = __builtin_amdgcn_wmma_f32_16x16x32_f16(false, a1, false, b, (short)0, acc[1][3], false, false);
  }
  int mb = (lane < 16) ? 0 : 8;
#pragma unroll
  for (int nt = 0; nt < 4; ++nt) {
    int n = n0 + nt * 16 + m;
    float bi = bias ? bias[n] : 0.f;
    float sc = scale ? scale[n] * scale_mul : 1.f;
    float sh = shift ? shift[n] : 0.f;
#pragma unroll
    for (int mt = 0; mt < 2; ++mt) {
#pragma unroll
      for (int v = 0; v < 8; ++v) {
        long mm = (long)t0 + mt * 16 + mb + v;
        float xv = (acc[mt][nt][v] + bi) * sc + sh;
        xv = hsd_act(xv, act);
        if (res) xv += res[(size_t)mm * ldr + n];
        Y[(size_t)mm * ldy + n] = xv;
      }
    }
  }
}

// Repack conv weight OIHW (128,128,3,3) -> [tap][o][c] contiguous.
__global__ void hsd_convw_repack_kernel(const float* __restrict__ w,
                                        float* __restrict__ out) {
  long total = (long)128 * 128 * 9;
  long i = (long)blockIdx.x * blockDim.x + threadIdx.x;
  if (i >= total) return;
  int tap = (int)(i % 9);
  int c = (int)((i / 9) % 128);
  int o = (int)(i / (9 * 128));
  out[(size_t)tap * 128 * 128 + (size_t)o * 128 + c] = w[i];
}

// ---------------------------------------------------------------------------
// Dense 3x3 conv (C=O=128) via implicit GEMM, 32 pixels x 64 out-channels per
// wave, repacked weights, clamped+predicated boundary loads (no divergent
// branches). Epilogue: BN scale/shift, optional residual, optional ReLU.
// ---------------------------------------------------------------------------
__global__ void hsd_conv3x3_kernel(const float* __restrict__ X,
                                   const float* __restrict__ W9,
                                   float* __restrict__ Y,
                                   int Bn, int H, int Wd,
                                   const float* __restrict__ scale,
                                   float scale_mul,
                                   const float* __restrict__ shift,
                                   const float* __restrict__ res, int act) {
  const int C = 128;
  int lane = threadIdx.x;
  int m = lane & 15;
  int kg = (lane < 16) ? 0 : 8;
  int n0 = blockIdx.x * 64;
  int t0 = blockIdx.y * 32;
  int xq[2], yq[2], bq[2];
#pragma unroll
  for (int mt = 0; mt < 2; ++mt) {
    long t = (long)t0 + mt * 16 + m;
    xq[mt] = (int)(t % Wd);
    yq[mt] = (int)((t / Wd) % H);
    bq[mt] = (int)(t / ((long)Wd * H));
  }
  v8f acc[2][4] = {};
#pragma unroll
  for (int tap = 0; tap < 9; ++tap) {
    int dy = tap / 3 - 1, dx = tap % 3 - 1;
    const float* apt[2];
    bool val[2];
#pragma unroll
    for (int mt = 0; mt < 2; ++mt) {
      int yy = yq[mt] + dy, xx = xq[mt] + dx;
      val[mt] = (yy >= 0) && (yy < H) && (xx >= 0) && (xx < Wd);
      int yc = yy < 0 ? 0 : (yy >= H ? H - 1 : yy);
      int xc = xx < 0 ? 0 : (xx >= Wd ? Wd - 1 : xx);
      apt[mt] = X + ((size_t)((bq[mt] * H + yc) * (long)Wd + xc)) * C;
    }
    const float* wt = W9 + (size_t)tap * C * 128;
    for (int k0 = 0; k0 < C; k0 += 32) {
      v16h a0 = hsd_load_fast_pred(apt[0] + k0, kg, val[0]);
      v16h a1 = hsd_load_fast_pred(apt[1] + k0, kg, val[1]);
#pragma unroll
      for (int nt = 0; nt < 4; ++nt) {
        v16h b = hsd_load_fast(wt + (size_t)(n0 + nt * 16 + m) * C + k0, kg);
        acc[0][nt] = __builtin_amdgcn_wmma_f32_16x16x32_f16(false, a0, false, b, (short)0, acc[0][nt], false, false);
        acc[1][nt] = __builtin_amdgcn_wmma_f32_16x16x32_f16(false, a1, false, b, (short)0, acc[1][nt], false, false);
      }
    }
  }
  int mb = (lane < 16) ? 0 : 8;
#pragma unroll
  for (int nt = 0; nt < 4; ++nt) {
    int n = n0 + nt * 16 + m;
    float sc = scale ? scale[n] * scale_mul : 1.f;
    float sh = shift ? shift[n] : 0.f;
#pragma unroll
    for (int mt = 0; mt < 2; ++mt) {
#pragma unroll
      for (int v = 0; v < 8; ++v) {
        long mm = (long)t0 + mt * 16 + mb + v;
        float xv = acc[mt][nt][v] * sc + sh;
        if (res) xv += res[(size_t)mm * C + n];
        if (act == ACT_RELU) xv = xv > 0.f ? xv : 0.f;
        Y[(size_t)mm * C + n] = xv;
      }
    }
  }
}

// ---------------------------------------------------------------------------
// LayerNorm over last dim (one thread per token). Optional SiLU-gate multiply.
// ---------------------------------------------------------------------------
__global__ void hsd_layernorm_kernel(const float* __restrict__ X, int ldx,
                                     const float* __restrict__ g,
                                     const float* __restrict__ b,
                                     float* __restrict__ Y, int ldy,
                                     long T, int C,
                                     const float* __restrict__ Z, int ldz) {
  long t = (long)blockIdx.x * blockDim.x + threadIdx.x;
  if (t >= T) return;
  const float* x = X + (size_t)t * ldx;
  float s = 0.f, s2 = 0.f;
  for (int c = 0; c < C; ++c) { float v = x[c]; s += v; s2 += v * v; }
  float mu = s / C;
  float var = s2 / C - mu * mu;
  float r = rsqrtf(var + 1e-5f);
  float* y = Y + (size_t)t * ldy;
  const float* z = Z ? Z + (size_t)t * ldz : nullptr;
  for (int c = 0; c < C; ++c) {
    float v = (x[c] - mu) * r * g[c] + b[c];
    if (z) { float zz = z[c]; v *= zz * (1.f / (1.f + __expf(-zz))); }
    y[c] = v;
  }
}

// Depthwise 3x3 + SiLU, channel-last. Input = first 256 channels of xz (ld 512).
__global__ void hsd_dwconv_silu_kernel(const float* __restrict__ X, int ldx,
                                       const float* __restrict__ Wt,
                                       const float* __restrict__ bias,
                                       float* __restrict__ Y, int ldy,
                                       int Bn, int H, int Wd, int C) {
  long total = (long)Bn * H * Wd * C;
  long i = (long)blockIdx.x * blockDim.x + threadIdx.x;
  if (i >= total) return;
  int c = (int)(i % C);
  long t = i / C;
  int x = (int)(t % Wd);
  int y = (int)((t / Wd) % H);
  int b = (int)(t / ((long)Wd * H));
  float acc = bias[c];
#pragma unroll
  for (int ky = 0; ky < 3; ++ky) {
#pragma unroll
    for (int kx = 0; kx < 3; ++kx) {
      int yy = y + ky - 1, xx = x + kx - 1;
      if (yy >= 0 && yy < H && xx >= 0 && xx < Wd) {
        acc += X[((size_t)((b * H + yy) * (long)Wd + xx)) * ldx + c] *
               Wt[c * 9 + ky * 3 + kx];
      }
    }
  }
  Y[(size_t)t * ldy + c] = acc * (1.f / (1.f + __expf(-acc)));
}

// xt[b, x*H+y, c] = xc[b, y*W+x, c]  (C = 256)
__global__ void hsd_transpose_tok_kernel(const float* __restrict__ in,
                                         float* __restrict__ out,
                                         int Bn, int H, int Wd) {
  long total = (long)Bn * H * Wd * 256;
  long i = (long)blockIdx.x * blockDim.x + threadIdx.x;
  if (i >= total) return;
  int c = (int)(i & 255);
  long t = i >> 8;
  long x = t % Wd;
  long y = (t / Wd) % H;
  long b = t / ((long)Wd * H);
  long tc = b * (long)H * Wd + x * H + y;
  out[(size_t)tc * 256 + c] = in[(size_t)t * 256 + c];
}

// ---------------------------------------------------------------------------
// Selective scan. Grid = Bn*4 blocks (one per (b, direction)); 256 threads = c.
// Directions 0/1 forward, 2/3 backward over the naturally ordered per-token
// projections (reversal cancels against the output un-reverse). B/C vectors
// staged through LDS.
// ---------------------------------------------------------------------------
__global__ void hsd_scan_kernel(const float* __restrict__ xc,
                                const float* __restrict__ xt,
                                const float* __restrict__ dbl,
                                const float* __restrict__ dlt,
                                const float* __restrict__ A_logs,
                                const float* __restrict__ Ds,
                                float* __restrict__ ys,
                                int Bn, int H, int Wd) {
  int k = blockIdx.x & 3;
  int b = blockIdx.x >> 2;
  int c = threadIdx.x;
  long L = (long)H * Wd;
  long T = (long)Bn * L;
  const float* u  = (k & 1) ? xt : xc;
  const float* dk = dbl + (size_t)k * T * 40;
  const float* dl = dlt + (size_t)k * T * 256;
  float* yk       = ys  + (size_t)k * T * 256;
  float A[16];
#pragma unroll
  for (int n = 0; n < 16; ++n)
    A[n] = -__expf(A_logs[(size_t)(k * 256 + c) * 16 + n]);
  float Dv = Ds[k * 256 + c];
  float h[16];
#pragma unroll
  for (int n = 0; n < 16; ++n) h[n] = 0.f;
  __shared__ float sB[16], sC[16];
  bool fwd = (k < 2);
  for (long s = 0; s < L; ++s) {
    long l = fwd ? s : (L - 1 - s);
    long t = b * L + l;
    __syncthreads();
    if (threadIdx.x < 32) {
      float v = dk[(size_t)t * 40 + 8 + threadIdx.x];
      if (threadIdx.x < 16) sB[threadIdx.x] = v; else sC[threadIdx.x - 16] = v;
    }
    __syncthreads();
    float dt = dl[(size_t)t * 256 + c];     // softplus already applied
    float uu = u[(size_t)t * 256 + c];
    float du = dt * uu;
    float y = uu * Dv;
#pragma unroll
    for (int n = 0; n < 16; ++n) {
      h[n] = h[n] * __expf(dt * A[n]) + du * sB[n];
      y += h[n] * sC[n];
    }
    yk[(size_t)t * 256 + c] = y;
  }
}

// y[t] = ys0[t] + ys2[t] + ys1[tc] + ys3[tc]   (tc = column-major index of t)
__global__ void hsd_combine_kernel(const float* __restrict__ ys,
                                   float* __restrict__ y,
                                   int Bn, int H, int Wd) {
  long T = (long)Bn * H * Wd;
  long total = T * 256;
  long i = (long)blockIdx.x * blockDim.x + threadIdx.x;
  if (i >= total) return;
  int c = (int)(i & 255);
  long t = i >> 8;
  long x = t % Wd;
  long yq = (t / Wd) % H;
  long b = t / ((long)Wd * H);
  long tc = b * (long)H * Wd + x * H + yq;
  size_t S = (size_t)T * 256;
  y[i] = ys[(size_t)t * 256 + c] + ys[2 * S + (size_t)t * 256 + c] +
         ys[S + (size_t)tc * 256 + c] + ys[3 * S + (size_t)tc * 256 + c];
}

// Bilinear 2x upsample (half-pixel centers, edge clamp) + skip add.
__global__ void hsd_upsample_add_kernel(const float* __restrict__ in,
                                        const float* __restrict__ skip,
                                        float* __restrict__ out,
                                        int Bn, int Hi, int Wi, int Ho, int Wo,
                                        int C) {
  long total = (long)Bn * Ho * Wo * C;
  long i = (long)blockIdx.x * blockDim.x + threadIdx.x;
  if (i >= total) return;
  int c = (int)(i % C);
  long t = i / C;
  int x = (int)(t % Wo);
  int y = (int)((t / Wo) % Ho);
  int b = (int)(t / ((long)Wo * Ho));
  float sy = (y + 0.5f) * ((float)Hi / Ho) - 0.5f;
  float sx = (x + 0.5f) * ((float)Wi / Wo) - 0.5f;
  float fy0 = floorf(sy), fx0 = floorf(sx);
  float wy = sy - fy0, wx = sx - fx0;
  int y0 = (int)fy0, x0 = (int)fx0;
  int y1 = y0 + 1, x1 = x0 + 1;
  y0 = y0 < 0 ? 0 : (y0 >= Hi ? Hi - 1 : y0);
  y1 = y1 < 0 ? 0 : (y1 >= Hi ? Hi - 1 : y1);
  x0 = x0 < 0 ? 0 : (x0 >= Wi ? Wi - 1 : x0);
  x1 = x1 < 0 ? 0 : (x1 >= Wi ? Wi - 1 : x1);
  size_t base = (size_t)b * Hi * Wi;
  float v00 = in[(base + (size_t)y0 * Wi + x0) * C + c];
  float v01 = in[(base + (size_t)y0 * Wi + x1) * C + c];
  float v10 = in[(base + (size_t)y1 * Wi + x0) * C + c];
  float v11 = in[(base + (size_t)y1 * Wi + x1) * C + c];
  float v = (1.f - wy) * ((1.f - wx) * v00 + wx * v01) +
            wy * ((1.f - wx) * v10 + wx * v11);
  out[i] = v + skip[i];
}

__global__ void hsd_nchw_to_nhwc_kernel(const float* __restrict__ in,
                                        float* __restrict__ out,
                                        int Bn, int C, int H, int Wd) {
  long total = (long)Bn * C * H * Wd;
  long i = (long)blockIdx.x * blockDim.x + threadIdx.x;
  if (i >= total) return;
  int w = (int)(i % Wd);
  int h = (int)((i / Wd) % H);
  int c = (int)((i / ((long)Wd * H)) % C);
  int b = (int)(i / ((long)Wd * H * C));
  out[((size_t)((b * H + h) * (long)Wd + w)) * C + c] = in[i];
}

__global__ void hsd_nhwc_to_nchw_kernel(const float* __restrict__ in,
                                        float* __restrict__ out,
                                        int Bn, int C, int H, int Wd) {
  long total = (long)Bn * C * H * Wd;
  long i = (long)blockIdx.x * blockDim.x + threadIdx.x;
  if (i >= total) return;
  int w = (int)(i % Wd);
  int h = (int)((i / Wd) % H);
  int c = (int)((i / ((long)Wd * H)) % C);
  int b = (int)(i / ((long)Wd * H * C));
  out[i] = in[((size_t)((b * H + h) * (long)Wd + w)) * C + c];
}

// ---------------------------------------------------------------------------
// Host orchestration
// ---------------------------------------------------------------------------
extern "C" void kernel_launch(void* const* d_in, const int* in_sizes, int n_in,
                              void* d_out, int out_size, void* d_ws, size_t ws_size,
                              hipStream_t stream) {
  (void)in_sizes; (void)n_in; (void)out_size; (void)ws_size;
  auto F = [&](int i) { return (const float*)d_in[i]; };

  const float* f0 = F(0);   // (4,96,64,64)
  const float* f1 = F(1);   // (4,192,32,32)
  const float* f2 = F(2);   // (4,384,16,16)
  const float* f3 = F(3);   // (4,768,8,8)
  const float* stem_w = F(4);
  const float* stem_b = F(5);
  const int VSS = 6;               // 4 blocks x 19 leaves
  const int TRN = VSS + 4 * 19;    // 3 x {w,b,g,bb}
  const int SMT = TRN + 3 * 4;     // 4 x {w1,g1,b1,w2,g2,b2}
  auto vp = [&](int i, int j) { return F(VSS + i * 19 + j); };
  auto tp = [&](int j, int k) { return F(TRN + j * 4 + k); };
  auto sp = [&](int s, int k) { return F(SMT + s * 6 + k); };

  const float BNS = 0.99999500003749973f;  // 1/sqrt(1 + 1e-5)
  const int Bn = 4;
  const size_t TM = 16384;

  float* wsf = (float*)d_ws;
  size_t off = 0;
  auto alloc = [&](size_t n) { float* p = wsf + off; off += n; return p; };
  float* bufA  = alloc(TM * 128);
  float* bufB  = alloc(TM * 128);
  float* bufC  = alloc(TM * 128);
  float* bufHN = alloc(TM * 128);
  float* bufXZ = alloc(TM * 512);        // also MLP hidden (reused)
  float* bufXC = alloc(TM * 256);
  float* bufXT = alloc(TM * 256);
  float* bufDBL = alloc((size_t)4 * TM * 40);
  float* bufDLT = alloc((size_t)4 * TM * 256);  // also NHWC staging (reused)
  float* bufYS  = alloc((size_t)4 * TM * 256);
  float* bufY   = alloc(TM * 256);       // also resblock tmp (reused)
  float* bufYN  = alloc(TM * 256);
  float* bufCW  = alloc((size_t)9 * 128 * 128);  // repacked conv weights
  float* bufST  = bufDLT;                // staging alias
  float* bufM   = bufXZ;                 // MLP hidden alias
  float* bufT   = bufY;                  // resblock tmp alias

  auto gemm = [&](const float* X, int ldx, const float* Wt, int ldw, float* Y,
                  int ldy, long T, int N, int K, const float* bias,
                  const float* scale, float smul, const float* shift,
                  const float* res, int ldr, int act) {
    if ((N % 64 == 0) && (K % 32 == 0) && (T % 32 == 0)) {
      dim3 g(N / 64, (unsigned)(T / 32));
      hsd_gemm_big_kernel<<<g, 32, 0, stream>>>(X, ldx, Wt, ldw, Y, ldy, (int)T,
                                                N, K, bias, scale, smul, shift,
                                                res, ldr, act);
    } else {
      dim3 g((N + 15) / 16, (unsigned)((T + 15) / 16));
      hsd_gemm_nt1_kernel<<<g, 32, 0, stream>>>(X, ldx, Wt, ldw, Y, ldy, (int)T,
                                                N, K, bias, scale, smul, shift,
                                                res, ldr, act);
    }
  };
  auto eltgrid = [](long total) { return dim3((unsigned)((total + 255) / 256)); };
  auto conv3 = [&](const float* X, const float* Wraw, float* Y, int H, int Wd,
                   const float* scale, const float* shift, const float* res,
                   int act) {
    hsd_convw_repack_kernel<<<eltgrid((long)9 * 128 * 128), 256, 0, stream>>>(
        Wraw, bufCW);
    long T = (long)Bn * H * Wd;
    dim3 g(2, (unsigned)(T / 32));
    hsd_conv3x3_kernel<<<g, 32, 0, stream>>>(X, bufCW, Y, Bn, H, Wd, scale, BNS,
                                             shift, res, act);
  };

  auto run_resblock = [&](int s, const float* in, float* tmp, float* out, int H,
                          int Wd) {
    conv3(in, sp(s, 0), tmp, H, Wd, sp(s, 1), sp(s, 2), nullptr, ACT_RELU);
    conv3(tmp, sp(s, 3), out, H, Wd, sp(s, 4), sp(s, 5), in, ACT_RELU);
  };

  auto run_vss = [&](int i, float* in, float* mid, float* out, int H, int Wd) {
    long T = (long)Bn * H * Wd;
    // ln1 -> in_proj (128 -> 512)
    hsd_layernorm_kernel<<<eltgrid(T), 256, 0, stream>>>(
        in, 128, vp(i, 0), vp(i, 1), bufHN, 128, T, 128, nullptr, 0);
    gemm(bufHN, 128, vp(i, 2), 128, bufXZ, 512, T, 512, 128, nullptr, nullptr,
         1.f, nullptr, nullptr, 0, ACT_NONE);
    // depthwise conv + SiLU on first 256 channels
    hsd_dwconv_silu_kernel<<<eltgrid(T * 256), 256, 0, stream>>>(
        bufXZ, 512, vp(i, 3), vp(i, 4), bufXC, 256, Bn, H, Wd, 256);
    hsd_transpose_tok_kernel<<<eltgrid(T * 256), 256, 0, stream>>>(bufXC, bufXT,
                                                                   Bn, H, Wd);
    // per-direction projections (natural token order; reversal handled in scan)
    for (int k = 0; k < 4; ++k) {
      const float* xk = (k & 1) ? bufXT : bufXC;
      gemm(xk, 256, vp(i, 5) + (size_t)k * 40 * 256, 256,
           bufDBL + (size_t)k * T * 40, 40, T, 40, 256, nullptr, nullptr, 1.f,
           nullptr, nullptr, 0, ACT_NONE);
      gemm(bufDBL + (size_t)k * T * 40, 40, vp(i, 6) + (size_t)k * 256 * 8, 8,
           bufDLT + (size_t)k * T * 256, 256, T, 256, 8, vp(i, 7) + k * 256,
           nullptr, 1.f, nullptr, nullptr, 0, ACT_SOFTPLUS);
    }
    hsd_scan_kernel<<<4 * Bn, 256, 0, stream>>>(bufXC, bufXT, bufDBL, bufDLT,
                                                vp(i, 8), vp(i, 9), bufYS, Bn, H,
                                                Wd);
    hsd_combine_kernel<<<eltgrid(T * 256), 256, 0, stream>>>(bufYS, bufY, Bn, H,
                                                             Wd);
    // out_norm * silu(z) -> out_proj + residual
    hsd_layernorm_kernel<<<eltgrid(T), 256, 0, stream>>>(
        bufY, 256, vp(i, 10), vp(i, 11), bufYN, 256, T, 256, bufXZ + 256, 512);
    gemm(bufYN, 256, vp(i, 12), 256, mid, 128, T, 128, 256, nullptr, nullptr,
         1.f, nullptr, in, 128, ACT_NONE);
    // MLP
    hsd_layernorm_kernel<<<eltgrid(T), 256, 0, stream>>>(
        mid, 128, vp(i, 13), vp(i, 14), bufHN, 128, T, 128, nullptr, 0);
    gemm(bufHN, 128, vp(i, 15), 128, bufM, 512, T, 512, 128, vp(i, 16), nullptr,
         1.f, nullptr, nullptr, 0, ACT_GELU);
    gemm(bufM, 512, vp(i, 17), 512, out, 128, T, 128, 512, vp(i, 18), nullptr,
         1.f, nullptr, mid, 128, ACT_NONE);
  };

  // ----- Stage 0: f3 (768, 8x8) -> stem -> VSS -----
  int H = 8, Wd = 8;
  long T = (long)Bn * H * Wd;
  hsd_nchw_to_nhwc_kernel<<<eltgrid((long)Bn * 768 * H * Wd), 256, 0, stream>>>(
      f3, bufST, Bn, 768, H, Wd);
  gemm(bufST, 768, stem_w, 768, bufA, 128, T, 128, 768, stem_b, nullptr, 1.f,
       nullptr, nullptr, 0, ACT_NONE);
  run_vss(0, bufA, bufB, bufA, H, Wd);

  // ----- Stages 1..3 -----
  const float* fins[3] = {f2, f1, f0};
  const int cins[3] = {384, 192, 96};
  for (int st = 1; st <= 3; ++st) {
    int Ho = H * 2, Wo = Wd * 2;
    long To = (long)Bn * Ho * Wo;
    int Ci = cins[st - 1];
    hsd_nchw_to_nhwc_kernel<<<eltgrid((long)Bn * Ci * Ho * Wo), 256, 0, stream>>>(
        fins[st - 1], bufST, Bn, Ci, Ho, Wo);
    // transition: conv1x1 + BN + ReLU (fused epilogue)
    gemm(bufST, Ci, tp(st - 1, 0), Ci, bufC, 128, To, 128, Ci, tp(st - 1, 1),
         tp(st - 1, 2), BNS, tp(st - 1, 3), nullptr, 0, ACT_RELU);
    hsd_upsample_add_kernel<<<eltgrid(To * 128), 256, 0, stream>>>(
        bufA, bufC, bufB, Bn, H, Wd, Ho, Wo, 128);
    run_resblock(st - 1, bufB, bufT, bufA, Ho, Wo);
    run_vss(st, bufA, bufB, bufA, Ho, Wo);
    H = Ho; Wd = Wo; T = To;
  }

  // ----- Final smooth resblock -> NCHW output -----
  run_resblock(3, bufA, bufT, bufB, H, Wd);
  hsd_nhwc_to_nchw_kernel<<<eltgrid((long)Bn * 128 * H * Wd), 256, 0, stream>>>(
      bufB, (float*)d_out, Bn, 128, H, Wd);
}
